// PartialFC_27462020890715
// MI455X (gfx1250) — matrix operand order repaired and checked by reading, not compiled, not run on hardware
//
#include <hip/hip_runtime.h>
#include <hip/hip_bf16.h>

// ---------------- problem constants ----------------
#define BATCH 512
#define DIM   512
#define NCLS  100000
#define NT    ((NCLS + 127) / 128)   // 782 class tiles of 128
#define SCALE 64.0f
#define COS_M 0.8775825618903728f    // cos(0.5)
#define SIN_M 0.479425538604203f     // sin(0.5)
#define LOG_CLAMP 69.07755278982137f // -log(1e-30)

typedef __attribute__((ext_vector_type(16))) __bf16        v16bf;
typedef __attribute__((ext_vector_type(8)))  float         v8f;
typedef __attribute__((ext_vector_type(8)))  unsigned int  v8u;

__device__ __forceinline__ unsigned int pack2bf(float a, float b) {
    // native v_cvt_pk_bf16_f32
    unsigned short lo = __builtin_bit_cast(unsigned short, (__bf16)a);
    unsigned short hi = __builtin_bit_cast(unsigned short, (__bf16)b);
    return (unsigned int)lo | ((unsigned int)hi << 16);
}

__device__ __forceinline__ v16bf frag_from(uint4 lo, uint4 hi) {
    v8u r;
    r[0] = lo.x; r[1] = lo.y; r[2] = lo.z; r[3] = lo.w;
    r[4] = hi.x; r[5] = hi.y; r[6] = hi.z; r[7] = hi.w;
    return __builtin_bit_cast(v16bf, r);   // pure register bit-cast
}

// unconditional 16-float load of one weight K-phase (64B, 16B-aligned chunks)
__device__ __forceinline__ void wload(const float* p,
                                      float4& a, float4& b, float4& c, float4& d) {
    const float4* q = (const float4*)p;
    a = q[0]; b = q[1]; c = q[2]; d = q[3];
}

// convert 16 f32 -> bf16, store 32B to LDS, return sum of squares
__device__ __forceinline__ float wstore(unsigned short* dst,
                                        float4 a, float4 b, float4 c, float4 d) {
    float ss = a.x*a.x + a.y*a.y + a.z*a.z + a.w*a.w
             + b.x*b.x + b.y*b.y + b.z*b.z + b.w*b.w
             + c.x*c.x + c.y*c.y + c.z*c.z + c.w*c.w
             + d.x*d.x + d.y*d.y + d.z*d.z + d.w*d.w;
    uint4 d0 = make_uint4(pack2bf(a.x, a.y), pack2bf(a.z, a.w),
                          pack2bf(b.x, b.y), pack2bf(b.z, b.w));
    uint4 d1 = make_uint4(pack2bf(c.x, c.y), pack2bf(c.z, c.w),
                          pack2bf(d.x, d.y), pack2bf(d.z, d.w));
    ((uint4*)dst)[0] = d0;
    ((uint4*)dst)[1] = d1;
    return ss;
}

// ---------------- kernel 1: normalize embeddings -> bf16 ----------------
__global__ void pfc_norm_emb(const float* __restrict__ E,
                             unsigned short* __restrict__ neb) {
    const int r   = blockIdx.x;        // one row per block
    const int tid = threadIdx.x;       // 256 threads
    const float* row = E + (size_t)r * DIM;
    float a = row[tid];
    float b = row[tid + 256];
    float ss = a * a + b * b;
    #pragma unroll
    for (int m = 16; m >= 1; m >>= 1) ss += __shfl_xor(ss, m, 32);
    __shared__ float wpart[8];
    const int lane = tid & 31, wave = tid >> 5;
    if (lane == 0) wpart[wave] = ss;
    __syncthreads();
    __shared__ float rn_s;
    if (tid == 0) {
        float t = 0.f;
        #pragma unroll
        for (int i = 0; i < 8; ++i) t += wpart[i];
        rn_s = rsqrtf(fmaxf(t, 1e-24f));
    }
    __syncthreads();
    const float rn = rn_s;
    neb[(size_t)r * DIM + tid]       = __builtin_bit_cast(unsigned short, (__bf16)(a * rn));
    neb[(size_t)r * DIM + tid + 256] = __builtin_bit_cast(unsigned short, (__bf16)(b * rn));
}

// ---------------- kernel 2: fused GEMM + norm + margin + tile softmax ----------------
__global__ __launch_bounds__(256)
void pfc_gemm_tile(const unsigned short* __restrict__ neb,  // [BATCH][DIM] bf16, L2-normalized
                   const float* __restrict__ W,             // [NCLS][DIM] f32, raw
                   const int* __restrict__ labels,          // [BATCH]
                   float* __restrict__ tgtz,                // [BATCH]
                   float* __restrict__ pmax,                // [NT][BATCH]
                   float* __restrict__ psum) {              // [NT][BATCH]
    const int tid    = threadIdx.x;
    const int ctile  = blockIdx.y;           // slow dim: class tile (weight reuse in L2
    const int btbase = blockIdx.x * 128;     // fast dim: 4 batch tiles run concurrently)
    const int ctbase = ctile * 128;

    const int lane    = tid & 31;
    const int wave    = tid >> 5;
    const int waveM   = wave >> 2;   // 0..1 : 64 batch rows each
    const int waveN   = wave & 3;    // 0..3 : 32 class cols each
    const int laneLow = lane & 15;
    const int laneHi  = lane >> 4;

    __shared__ unsigned short sB[2][128 * 32]; // weight K-chunk, bf16, n-major
    __shared__ float s_ss[256];
    __shared__ float s_rn[128];
    __shared__ int   s_lab[128];
    __shared__ float s_pm[128][4];
    __shared__ float s_ps[128][4];

    if (tid < 128) s_lab[tid] = labels[btbase + tid];

    // staging mapping: thread -> (class row n = tid>>1, k phase = (tid&1)*16).
    // Out-of-range rows are clamped to the last class: their GEMM results and
    // norms are dead (epilogue masks cg >= NCLS), so no branch is needed.
    const int sn    = tid >> 1;
    const int skoff = (tid & 1) * 16;
    const int scl   = min(ctbase + sn, NCLS - 1);
    const float* wbase = W + (size_t)scl * DIM + skoff;
    unsigned short* sdst0 = &sB[0][sn * 32 + skoff];
    unsigned short* sdst1 = &sB[1][sn * 32 + skoff];

    float ss = 0.f;            // per-thread sum of squares of staged weights
    float4 pf0, pf1, pf2, pf3; // prefetch regs: 16 f32

    v8f acc[4][2] = {};        // 4 M-tiles x 2 N-tiles of 16x16 f32
    const int arow = btbase + waveM * 64 + laneLow;

    wload(wbase, pf0, pf1, pf2, pf3);
    ss += wstore(sdst0, pf0, pf1, pf2, pf3);

    for (int kc = 0; kc < DIM / 32; ++kc) {
        __syncthreads();
        const bool more = (kc + 1) < DIM / 32;
        if (more) wload(wbase + (kc + 1) * 32, pf0, pf1, pf2, pf3);

        const int buf = kc & 1;
        const int kb  = kc * 32 + laneHi * 16;  // A: lanes 0-15 get K 0..15, 16-31 get 16..31

        v16bf av[4];
        #pragma unroll
        for (int mt = 0; mt < 4; ++mt) {
            const unsigned short* pa = neb + (size_t)(arow + mt * 16) * DIM + kb;
            av[mt] = frag_from(*(const uint4*)pa, *(const uint4*)(pa + 8));
        }
        v16bf bv[2];
        #pragma unroll
        for (int nt = 0; nt < 2; ++nt) {
            // 32B-aligned contiguous 16 K values for this column
            const v8u* pb = (const v8u*)&sB[buf][(waveN * 32 + nt * 16 + laneLow) * 32 + laneHi * 16];
            bv[nt] = __builtin_bit_cast(v16bf, *pb);
        }
        #pragma unroll
        for (int mt = 0; mt < 4; ++mt)
            #pragma unroll
            for (int nt = 0; nt < 2; ++nt)
                acc[mt][nt] = __builtin_amdgcn_wmma_f32_16x16x32_bf16(
                    false, av[mt], false, bv[nt], (short)0, acc[mt][nt], false, false);

        if (more) ss += wstore(buf ? sdst0 : sdst1, pf0, pf1, pf2, pf3);
    }

    // ---- class-row reciprocal norms from staged sum of squares ----
    s_ss[tid] = ss;
    __syncthreads();
    if (tid < 128) {
        float t = s_ss[2 * tid] + s_ss[2 * tid + 1];
        s_rn[tid] = rsqrtf(fmaxf(t, 1e-24f));
    }
    __syncthreads();

    const float rn0 = s_rn[waveN * 32 + laneLow];
    const float rn1 = s_rn[waveN * 32 + 16 + laneLow];

    // ---- epilogue: margin, scale, per-row tile-softmax partials ----
    #pragma unroll
    for (int mt = 0; mt < 4; ++mt) {
        #pragma unroll
        for (int i = 0; i < 8; ++i) {
            const int rloc = waveM * 64 + mt * 16 + laneHi * 8 + i;
            const int lbl  = s_lab[rloc];
            float z[2];
            #pragma unroll
            for (int nt = 0; nt < 2; ++nt) {
                const int cg = ctbase + waveN * 32 + nt * 16 + laneLow;
                float t = acc[mt][nt][i] * (nt ? rn1 : rn0);
                t = fminf(1.f, fmaxf(-1.f, t));
                float zz;
                if (cg >= NCLS) {
                    zz = -1e30f;
                } else if (cg == lbl) {
                    float s2 = fmaxf(0.f, 1.f - t * t);
                    float tm = t * COS_M - sqrtf(s2) * SIN_M;
                    zz = SCALE * tm;
                    tgtz[btbase + rloc] = zz;      // exactly one writer grid-wide
                } else {
                    zz = SCALE * t;
                }
                z[nt] = zz;
            }
            float rm = fmaxf(z[0], z[1]);
            #pragma unroll
            for (int m = 1; m <= 8; m <<= 1) rm = fmaxf(rm, __shfl_xor(rm, m, 32));
            float se = __expf(z[0] - rm) + __expf(z[1] - rm);
            #pragma unroll
            for (int m = 1; m <= 8; m <<= 1) se += __shfl_xor(se, m, 32);
            if (laneLow == 0) { s_pm[rloc][waveN] = rm; s_ps[rloc][waveN] = se; }
        }
    }
    __syncthreads();

    if (tid < 128) {
        float m = s_pm[tid][0];
        #pragma unroll
        for (int j = 1; j < 4; ++j) m = fmaxf(m, s_pm[tid][j]);
        float s = 0.f;
        #pragma unroll
        for (int j = 0; j < 4; ++j) s += s_ps[tid][j] * __expf(s_pm[tid][j] - m);
        pmax[(size_t)ctile * BATCH + btbase + tid] = m;
        psum[(size_t)ctile * BATCH + btbase + tid] = s;
    }
}

// ---------------- kernel 3a: per-row log-sum-exp merge over class tiles ----------------
__global__ void pfc_lse_row(const float* __restrict__ pmax,
                            const float* __restrict__ psum,
                            const float* __restrict__ tgtz,
                            float* __restrict__ loss_row) {
    const int r   = blockIdx.x;     // one batch row per block
    const int tid = threadIdx.x;    // 128 threads
    float m = -1e30f, s = 0.f;
    for (int t = tid; t < NT; t += 128) {
        float mt_ = pmax[(size_t)t * BATCH + r];
        float st_ = psum[(size_t)t * BATCH + r];
        float nm  = fmaxf(m, mt_);
        s = s * __expf(m - nm) + st_ * __expf(mt_ - nm);
        m = nm;
    }
    __shared__ float sm[128], sv[128];
    sm[tid] = m; sv[tid] = s;
    __syncthreads();
    for (int st = 64; st > 0; st >>= 1) {
        if (tid < st) {
            float m2 = sm[tid + st], s2 = sv[tid + st];
            float nm = fmaxf(sm[tid], m2);
            sv[tid] = sv[tid] * __expf(sm[tid] - nm) + s2 * __expf(m2 - nm);
            sm[tid] = nm;
        }
        __syncthreads();
    }
    if (tid == 0)
        loss_row[r] = fminf(sm[0] + __logf(sv[0]) - tgtz[r], LOG_CLAMP);
}

// ---------------- kernel 3b: mean over batch ----------------
__global__ void pfc_mean(const float* __restrict__ loss_row,
                         float* __restrict__ out) {
    const int r = threadIdx.x;   // 512 threads
    __shared__ float red[512];
    red[r] = loss_row[r];
    __syncthreads();
    for (int st = 256; st > 0; st >>= 1) {
        if (r < st) red[r] += red[r + st];
        __syncthreads();
    }
    if (r == 0) out[0] = red[0] * (1.f / (float)BATCH);
}

// ---------------- host launcher ----------------
extern "C" void kernel_launch(void* const* d_in, const int* in_sizes, int n_in,
                              void* d_out, int out_size, void* d_ws, size_t ws_size,
                              hipStream_t stream) {
    (void)in_sizes; (void)n_in; (void)out_size; (void)ws_size;
    const float* E = (const float*)d_in[0];
    const float* W = (const float*)d_in[1];
    const int*   L = (const int*)d_in[2];

    char* ws = (char*)d_ws;
    unsigned short* neb   = (unsigned short*)ws;                      // 512 KB bf16 embeddings
    float*          tgtz  = (float*)(ws + (size_t)BATCH * DIM * 2);   // 2 KB
    float*          lrow  = tgtz + BATCH;                             // 2 KB
    float*          pmax  = lrow + BATCH;
    float*          psum  = pmax + (size_t)NT * BATCH;

    pfc_norm_emb<<<BATCH, 256, 0, stream>>>(E, neb);
    // x = batch tile (fast-varying -> concurrent blocks share the same weight tile in L2)
    pfc_gemm_tile<<<dim3(BATCH / 128, NT), 256, 0, stream>>>(neb, W, L, tgtz, pmax, psum);
    pfc_lse_row<<<BATCH, 128, 0, stream>>>(pmax, psum, tgtz, lrow);
    pfc_mean<<<1, 512, 0, stream>>>(lrow, (float*)d_out);
}
